// UnifiedEmbed_8589934826
// MI455X (gfx1250) — matrix operand order repaired and checked by reading, not compiled, not run on hardware
//
#include <hip/hip_runtime.h>
#include <hip/hip_bf16.h>

typedef __bf16 bf16_t;
typedef __attribute__((ext_vector_type(16))) __bf16 v16bf;
typedef __attribute__((ext_vector_type(8)))  float  v8f;

#define TILE_M 256
#define TILE_N 128
#define TILE_K 32
#define LDT    56   // padded LDS row stride in bf16 (112 B = 7*16: 16B-aligned, conflict-friendly)

#if defined(__AMDGCN__) && __has_builtin(__builtin_amdgcn_global_load_async_to_lds_b128)
#define USE_ASYNC_LDS 1
// Builtin signature (from hipcc diagnostic): param0 = 'int __vector(4) __device__ *'
// i.e. address_space(1) pointer to a vector_size(16) int; param1 = LDS-side pointer.
typedef int v4i_gcc __attribute__((vector_size(16)));
typedef __attribute__((address_space(1))) v4i_gcc* gp128_t;  // global (device) int4*
typedef __attribute__((address_space(3))) v4i_gcc* sp128_t;  // LDS (shared) int4*
#else
#define USE_ASYNC_LDS 0
#endif

__device__ __forceinline__ void wait_async_le6() {
#if USE_ASYNC_LDS
#if __has_builtin(__builtin_amdgcn_s_wait_asynccnt)
  __builtin_amdgcn_s_wait_asynccnt(6);
#else
  asm volatile("s_wait_asynccnt 0x6" ::: "memory");
#endif
#endif
}
__device__ __forceinline__ void wait_async_le0() {
#if USE_ASYNC_LDS
#if __has_builtin(__builtin_amdgcn_s_wait_asynccnt)
  __builtin_amdgcn_s_wait_asynccnt(0);
#else
  asm volatile("s_wait_asynccnt 0x0" ::: "memory");
#endif
#endif
}

union FragBF   { v16bf v; unsigned int u[8]; };
union PackB128 { bf16_t h[8]; uint4 q; };

// A fragment (16 rows x 32 K) from LDS tile sA[m][k], row stride LDT bf16.
// ISA 7.12.2 16-bit A 16x32: lanes0-15 M=0..15, VGPR v -> K=(v>>2)*16+(v&3)*2 (+1);
// lanes16-31 same M, +8 within each 16-wide K group.
__device__ __forceinline__ v16bf load_frag_a(const bf16_t* sA, int mBase, int lane) {
  const int m = mBase + (lane & 15);
  const int half = lane >> 4;
  const unsigned int* row = (const unsigned int*)(sA + m * LDT); // uint = bf16 pair
  FragBF f;
#pragma unroll
  for (int v = 0; v < 8; ++v) {
    const int k2 = (v >> 2) * 8 + half * 4 + (v & 3);
    f.u[v] = row[k2];
  }
  return f.v;
}

// B fragment (32 K x 16 N) from LDS tile sB[n][k] (B^T layout), row stride LDT bf16.
// same K layout as A: VGPR v -> K=(v>>2)*16+half*8+(v&3)*2 (+1); N = lane&15.
__device__ __forceinline__ v16bf load_frag_b(const bf16_t* sB, int nBase, int lane) {
  const int n = nBase + (lane & 15);
  const int half = lane >> 4;
  const unsigned int* row = (const unsigned int*)(sB + n * LDT);
  FragBF f;
#pragma unroll
  for (int v = 0; v < 8; ++v) {
    f.u[v] = row[(v >> 2) * 8 + half * 4 + (v & 3)];
  }
  return f.v;
}

// X[m][k] = (bf16)tok_embed[idxs[m]][k] — gather+convert once; X (128 MB) then lives in L2.
__global__ __launch_bounds__(256)
void gather_convert_kernel(const int* __restrict__ idxs,
                           const float* __restrict__ tok_embed,
                           bf16_t* __restrict__ X) {
  const long long i = (long long)blockIdx.x * blockDim.x + threadIdx.x; // one per 8 elems
  const int m  = (int)(i >> 9);          // 512 threads per 4096-wide row
  const int kc = ((int)i & 511) * 8;
  const long long src = (long long)idxs[m] * 4096 + kc;
  const float4 v0 = *(const float4*)(tok_embed + src);
  const float4 v1 = *(const float4*)(tok_embed + src + 4);
  PackB128 pk;
  pk.h[0] = (bf16_t)v0.x; pk.h[1] = (bf16_t)v0.y;
  pk.h[2] = (bf16_t)v0.z; pk.h[3] = (bf16_t)v0.w;
  pk.h[4] = (bf16_t)v1.x; pk.h[5] = (bf16_t)v1.y;
  pk.h[6] = (bf16_t)v1.z; pk.h[7] = (bf16_t)v1.w;
  *(uint4*)(X + (long long)m * 4096 + kc) = pk.q;
}

// dst[n][k] = (bf16)src[k][n]  (row-major [N][K] output)
__global__ __launch_bounds__(256)
void convert_transpose_kernel(const float* __restrict__ src, bf16_t* __restrict__ dst,
                              int K, int N) {
  long long i = (long long)blockIdx.x * blockDim.x + threadIdx.x;
  if (i >= (long long)K * N) return;
  const int k = (int)(i % K);
  const int n = (int)(i / K);
  dst[(long long)n * K + k] = (bf16_t)src[(long long)k * N + n];
}

// C[m][n] = A[m][:] @ Bt[n][:] + bias[n]; optionally exact GELU + bf16 output.
// WG: 256 threads = 8 waves (4 in M x 2 in N), wave tile 64x64 -> WG tile 256x128.
// Async path: double-buffered LDS fed by GLOBAL_LOAD_ASYNC_TO_LDS_B128 (zero staging VGPRs).
template <int KDIM, bool GELU>
__global__ __launch_bounds__(256)
void gemm_bias_kernel(const bf16_t* __restrict__ A,   // [M][KDIM]
                      const bf16_t* __restrict__ Bt,  // [1024][KDIM]
                      const float*  __restrict__ bias,
                      bf16_t* __restrict__ outH,      // used when GELU
                      float*  __restrict__ outF) {    // used when !GELU
  const int tid  = threadIdx.x;
  const int lane = tid & 31;
  const int wave = tid >> 5;
  const int waveM = wave & 3;   // 4 waves x 64 rows = 256
  const int waveN = wave >> 2;  // 2 waves x 64 cols = 128

  const int m0 = blockIdx.x * TILE_M;
  const int n0 = blockIdx.y * TILE_N;

  // Per-thread staging geometry: chunk c = p*256 + tid -> row p*64 + tid/4, col (tid&3)*8.
  const int srow = tid >> 2;
  const int scol = (tid & 3) * 8;
  const bf16_t* aBase = A  + (long long)(m0 + srow) * KDIM + scol;
  const bf16_t* bBase = Bt + (long long)(n0 + srow) * KDIM + scol;

  const int mBase = waveM * 64;
  const int nBase = waveN * 64;

  v8f acc[4][4] = {};

#if USE_ASYNC_LDS
  __shared__ bf16_t sA[2][TILE_M * LDT];
  __shared__ bf16_t sB[2][TILE_N * LDT];
  bf16_t* sAp[2] = { &sA[0][srow * LDT + scol], &sA[1][srow * LDT + scol] };
  bf16_t* sBp[2] = { &sB[0][srow * LDT + scol], &sB[1][srow * LDT + scol] };

  // prologue: stage K-step 0 into buffer 0 (6 x async b128 per thread)
#pragma unroll
  for (int p = 0; p < 4; ++p)
    __builtin_amdgcn_global_load_async_to_lds_b128(
        (gp128_t)(aBase + (long long)p * 64 * KDIM),
        (sp128_t)(sAp[0] + p * 64 * LDT), 0, 0);
#pragma unroll
  for (int p = 0; p < 2; ++p)
    __builtin_amdgcn_global_load_async_to_lds_b128(
        (gp128_t)(bBase + (long long)p * 64 * KDIM),
        (sp128_t)(sBp[0] + p * 64 * LDT), 0, 0);

  int cur = 0;
  for (int k0 = 0; k0 < KDIM; k0 += TILE_K) {
    const int nxt = cur ^ 1;
    const int kn  = k0 + TILE_K;
    if (kn < KDIM) {
      // stage K-step i+1; overlaps with this step's WMMAs
#pragma unroll
      for (int p = 0; p < 4; ++p)
        __builtin_amdgcn_global_load_async_to_lds_b128(
            (gp128_t)(aBase + (long long)p * 64 * KDIM + kn),
            (sp128_t)(sAp[nxt] + p * 64 * LDT), 0, 0);
#pragma unroll
      for (int p = 0; p < 2; ++p)
        __builtin_amdgcn_global_load_async_to_lds_b128(
            (gp128_t)(bBase + (long long)p * 64 * KDIM + kn),
            (sp128_t)(sBp[nxt] + p * 64 * LDT), 0, 0);
      wait_async_le6();   // in-order completion: buffer `cur`'s 6 copies are done
    } else {
      wait_async_le0();
    }
    __syncthreads();      // every wave's copies into `cur` are complete

    const bf16_t* cA = sA[cur];
    const bf16_t* cB = sB[cur];
    v16bf a[4];
#pragma unroll
    for (int mi = 0; mi < 4; ++mi) a[mi] = load_frag_a(cA, mBase + mi * 16, lane);
#pragma unroll
    for (int ni = 0; ni < 4; ++ni) {
      const v16bf b = load_frag_b(cB, nBase + ni * 16, lane);
#pragma unroll
      for (int mi = 0; mi < 4; ++mi) {
        acc[mi][ni] = __builtin_amdgcn_wmma_f32_16x16x32_bf16(false, a[mi], false, b,
                                                              (short)0, acc[mi][ni],
                                                              false, false);
      }
    }
    __syncthreads();      // all reads of `cur` done before it is overwritten at step i+2
    cur = nxt;
  }
#else
  // Fallback: single-buffer staging through VGPRs (round-2 structure, no long live ranges)
  __shared__ bf16_t sA[TILE_M * LDT];
  __shared__ bf16_t sB[TILE_N * LDT];
  bf16_t* sAp = sA + srow * LDT + scol;
  bf16_t* sBp = sB + srow * LDT + scol;

  for (int k0 = 0; k0 < KDIM; k0 += TILE_K) {
    __syncthreads();
#pragma unroll
    for (int p = 0; p < 4; ++p)
      *(uint4*)(sAp + p * 64 * LDT) = *(const uint4*)(aBase + (long long)p * 64 * KDIM + k0);
#pragma unroll
    for (int p = 0; p < 2; ++p)
      *(uint4*)(sBp + p * 64 * LDT) = *(const uint4*)(bBase + (long long)p * 64 * KDIM + k0);
    __syncthreads();

    v16bf a[4];
#pragma unroll
    for (int mi = 0; mi < 4; ++mi) a[mi] = load_frag_a(sA, mBase + mi * 16, lane);
#pragma unroll
    for (int ni = 0; ni < 4; ++ni) {
      const v16bf b = load_frag_b(sB, nBase + ni * 16, lane);
#pragma unroll
      for (int mi = 0; mi < 4; ++mi) {
        acc[mi][ni] = __builtin_amdgcn_wmma_f32_16x16x32_bf16(false, a[mi], false, b,
                                                              (short)0, acc[mi][ni],
                                                              false, false);
      }
    }
  }
#endif

  // ---- epilogue: C/D layout (VGPR r: M = 8*halflane + r, N = lane&15) ----
  const int half = lane >> 4;
  const int nl   = lane & 15;
#pragma unroll
  for (int mi = 0; mi < 4; ++mi) {
#pragma unroll
    for (int ni = 0; ni < 4; ++ni) {
      const int gn = n0 + waveN * 64 + ni * 16 + nl;
      const float bv = bias[gn];
#pragma unroll
      for (int r = 0; r < 8; ++r) {
        const int gm = m0 + waveM * 64 + mi * 16 + half * 8 + r;
        const float x = acc[mi][ni][r] + bv;
        if (GELU) {
          const float g = 0.5f * x * (1.0f + erff(x * 0.70710678118654752f));
          outH[(long long)gm * 1024 + gn] = (bf16_t)g;
        } else {
          outF[(long long)gm * 1024 + gn] = x;
        }
      }
    }
  }
}

extern "C" void kernel_launch(void* const* d_in, const int* in_sizes, int n_in,
                              void* d_out, int out_size, void* d_ws, size_t ws_size,
                              hipStream_t stream) {
  const int*   idxs = (const int*)d_in[0];     // [4,4096]
  const float* tok  = (const float*)d_in[1];   // [50257,4096]
  const float* W1   = (const float*)d_in[2];   // [4096,1024]
  const float* b1   = (const float*)d_in[3];   // [1024]
  const float* W2   = (const float*)d_in[4];   // [1024,1024]
  const float* b2   = (const float*)d_in[5];   // [1024]
  float* out = (float*)d_out;                  // [4,4096,1024]

  char* ws = (char*)d_ws;
  bf16_t* X   = (bf16_t*)ws;                                   // 128 MB [16384][4096]
  bf16_t* w1t = (bf16_t*)(ws + (size_t)16384 * 4096 * 2);      //   8 MB [1024][4096]
  bf16_t* w2t = (bf16_t*)(ws + (size_t)16384 * 4096 * 2
                             + (size_t)4096 * 1024 * 2);       //   2 MB [1024][1024]
  bf16_t* H   = (bf16_t*)(ws + (size_t)16384 * 4096 * 2
                             + (size_t)4096 * 1024 * 2
                             + (size_t)1024 * 1024 * 2);       //  32 MB [16384][1024]

  // 1) gather + f32->bf16 once (random-row HBM reads happen exactly once)
  gather_convert_kernel<<<(16384LL * 512) / 256, 256, 0, stream>>>(idxs, tok, X);
  // 2) weight convert/transpose (tiny)
  convert_transpose_kernel<<<(4096 * 1024 + 255) / 256, 256, 0, stream>>>(W1, w1t, 4096, 1024);
  convert_transpose_kernel<<<(1024 * 1024 + 255) / 256, 256, 0, stream>>>(W2, w2t, 1024, 1024);

  dim3 grid(16384 / TILE_M, 1024 / TILE_N);   // 64 x 8
  // 3) H = GELU(X @ W1 + b1) in bf16
  gemm_bias_kernel<4096, true><<<grid, 256, 0, stream>>>(X, w1t, b1, H, nullptr);
  // 4) out = H @ W2 + b2 in f32
  gemm_bias_kernel<1024, false><<<grid, 256, 0, stream>>>(H, w2t, b2, nullptr, out);
}
